// MultiHeadedAttention_47863115546923
// MI455X (gfx1250) — matrix operand order repaired and checked by reading, not compiled, not run on hardware
//
#include <hip/hip_runtime.h>

typedef __bf16 bf16_t;
typedef __attribute__((ext_vector_type(16))) bf16_t v16bf;
typedef __attribute__((ext_vector_type(8)))  float  v8f;

union FragBF { v16bf v; unsigned u[8]; };

__device__ __forceinline__ unsigned short f2bf(float f) {
  unsigned u = __builtin_bit_cast(unsigned, f);
  u += 0x7FFFu + ((u >> 16) & 1u);          // round-to-nearest-even
  return (unsigned short)(u >> 16);
}
__device__ __forceinline__ float bf2f(unsigned short h) {
  return __builtin_bit_cast(float, (unsigned)h << 16);
}
__device__ __forceinline__ float mishf(float x) {
  float e  = __expf(x);
  float sp = (x > 20.f) ? x : log1pf(e);    // softplus
  return x * tanhf(sp);
}

// ---------------------------------------------------------------------------
// W[512,512] f32 (row k, col n)  ->  Wt[512,512] bf16 stored [n][k]
// 32x32 LDS tile transpose; grid (16,16), block 256 (32x8).
// ---------------------------------------------------------------------------
__global__ __launch_bounds__(256)
void convert_w_kernel(const float* __restrict__ W, unsigned short* __restrict__ Wt)
{
  __shared__ float tile[32][33];
  const int tx = threadIdx.x & 31, ty = threadIdx.x >> 5;
  const int k0 = blockIdx.x * 32, n0 = blockIdx.y * 32;
#pragma unroll
  for (int i = 0; i < 32; i += 8)
    tile[ty + i][tx] = W[(long)(k0 + ty + i) * 512 + n0 + tx];
  __syncthreads();
#pragma unroll
  for (int i = 0; i < 32; i += 8)
    Wt[(long)(n0 + ty + i) * 512 + k0 + tx] = f2bf(tile[tx][ty + i]);
}

// ---------------------------------------------------------------------------
// Y[M,512] = op( X[M,512] @ W[512,512] + bias )      (bf16 WMMA, f32 acc)
//   Wt: pre-converted bf16 weights, [n][k] layout -> B staging is a raw copy
//   SCALE_MSG: A row built on the fly as msg[b,h,m,n] * kproj_bf16[row, k]
//   MISH:      apply mish epilogue
//   OUT_BF16:  store bf16 to outBF instead of f32 to outF
// Block: 256 threads (8 waves). Tile: 64 rows x 512 cols, K-chunk = 32.
// Wave w: m-block = (w&3)*16, n-half = (w>>2)*256 -> 16 n-tiles of 16.
// B fragments double-buffered in registers so DS loads overlap WMMA.
// ---------------------------------------------------------------------------
template<bool SCALE_MSG, bool MISH, bool OUT_BF16>
__global__ __launch_bounds__(256)
void gemm512_wmma(const float* __restrict__ Af32,
                  const unsigned short* __restrict__ Abf,
                  const float* __restrict__ msg,
                  const unsigned short* __restrict__ Wt,
                  const float* __restrict__ bias,
                  float* __restrict__ outF,
                  unsigned short* __restrict__ outBF,
                  int M)
{
  __shared__ unsigned short As[64 * 40];    // [r][k], stride 40 (pad)
  __shared__ unsigned short Bs[512 * 40];   // [n][k], stride 40 (pad)

  const int tid  = threadIdx.x;
  const int wave = tid >> 5, lane = tid & 31;
  const int mb   = wave & 3, nhf = wave >> 2;
  const int ml   = lane & 15, hi = lane >> 4;
  const long rowBase = (long)blockIdx.x * 64;

  v8f acc[16];
#pragma unroll
  for (int t = 0; t < 16; ++t)
    acc[t] = (v8f){0.f, 0.f, 0.f, 0.f, 0.f, 0.f, 0.f, 0.f};

  for (int kc = 0; kc < 16; ++kc) {
    const int kbase = kc * 32;
    __syncthreads();

    // ---- stage A tile (64 x 32): one bf16 pair per packed b32 store ----
#pragma unroll
    for (int it = 0; it < 4; ++it) {
      int idx = it * 256 + tid;                // 0..1023
      int r = idx >> 4, kp = idx & 15;         // row, k-pair
      long g = rowBase + r;
      unsigned pk;
      if (SCALE_MSG) {
        int h = kbase >> 6;                    // head is constant per chunk
        unsigned two = (g < M) ? *(const unsigned*)&Abf[g * 512 + kbase + 2 * kp] : 0u;
        int n = (int)(g & 63), m = (int)((g >> 6) & 63), b = (int)(g >> 12);
        float mg = (g < M) ? msg[(((b * 8 + h) * 64 + m) << 6) + n] : 0.f;
        float a0 = bf2f((unsigned short)(two & 0xFFFFu)) * mg;
        float a1 = bf2f((unsigned short)(two >> 16)) * mg;
        pk = (unsigned)f2bf(a0) | ((unsigned)f2bf(a1) << 16);
      } else {
        float a0 = 0.f, a1 = 0.f;
        if (g < M) {
          float2 p = *(const float2*)&Af32[g * 512 + kbase + 2 * kp];
          a0 = p.x; a1 = p.y;
        }
        pk = (unsigned)f2bf(a0) | ((unsigned)f2bf(a1) << 16);
      }
      *(unsigned*)&As[r * 40 + 2 * kp] = pk;
    }

    // prefetch next K-chunk of Wt while staging this one
    if (kbase + 32 < 512)
      __builtin_prefetch(&Wt[(long)tid * 512 + kbase + 32], 0, 0);

    // ---- stage B tile: pure 16B copies of pre-transposed bf16 weights ----
#pragma unroll
    for (int it = 0; it < 8; ++it) {
      int idx = it * 256 + tid;                // 0..2047
      int n = idx >> 2, q = idx & 3;           // row n, 16B quarter q
      uint4 p = *(const uint4*)&Wt[(long)n * 512 + kbase + q * 8];
      *(uint4*)&Bs[n * 40 + q * 8] = p;
    }
    __syncthreads();

    // ---- A fragment: 16x32 bf16, ISA layout (lane ml = row; hi = K half) ----
    FragBF aF;
#pragma unroll
    for (int j = 0; j < 8; ++j) {
      int kj = 8 * hi + ((j < 4) ? (2 * j) : (16 + 2 * (j - 4)));
      aF.u[j] = *(const unsigned*)&As[(mb * 16 + ml) * 40 + kj];
    }

    // ---- 16 n-tiles, B fragment double-buffered: load t+1 before WMMA t ----
    FragBF bF[2];
    {
      const unsigned* bp = (const unsigned*)&Bs[(nhf * 256 + ml) * 40 + hi * 8];
#pragma unroll
      for (int j = 0; j < 8; ++j) bF[0].u[j] = bp[(j < 4) ? j : (j + 4)];
    }
#pragma unroll
    for (int t = 0; t < 16; ++t) {
      if (t < 15) {
        int ncol = nhf * 256 + (t + 1) * 16 + ml;
        const unsigned* bp = (const unsigned*)&Bs[ncol * 40 + hi * 8];
#pragma unroll
        for (int j = 0; j < 8; ++j) bF[(t + 1) & 1].u[j] = bp[(j < 4) ? j : (j + 4)];
      }
      acc[t] = __builtin_amdgcn_wmma_f32_16x16x32_bf16(
                 false, aF.v, false, bF[t & 1].v, (short)0, acc[t], false, false);
    }
  }

  // ---- epilogue: C layout = VGPR r -> M = r (lanes 0-15) / r+8 (lanes 16-31)
#pragma unroll
  for (int t = 0; t < 16; ++t) {
    int col = nhf * 256 + t * 16 + ml;
    float bc = bias[col];
#pragma unroll
    for (int r = 0; r < 8; ++r) {
      int mloc = mb * 16 + ((hi == 0) ? r : r + 8);
      long g = rowBase + mloc;
      if (g < M) {
        float val = acc[t][r] + bc;
        if (MISH) val = mishf(val);
        if (OUT_BF16) outBF[g * 512 + col] = f2bf(val);
        else          outF [g * 512 + col] = val;
      }
    }
  }
}

// ---------------------------------------------------------------------------
// One block per (b,h): scores (q.k, both directions), softmaxes, message
// (incl. diagonal correction), node_hidden. kproj is bf16 and L2-resident.
// ---------------------------------------------------------------------------
__global__ __launch_bounds__(256)
void attn_msg_kernel(const float* __restrict__ q,
                     const float* __restrict__ v,
                     const unsigned short* __restrict__ kbf,
                     float* __restrict__ msg_out,
                     float* __restrict__ nh_out)
{
  __shared__ float qs[64 * 64];
  __shared__ float so[64 * 64];   // out_scores -> out_attn -> message
  __shared__ float si[64 * 64];   // in_scores  -> in_attn
  __shared__ float dg[64];

  const int bh = blockIdx.x, b = bh >> 3, h = bh & 7;
  const int tid = threadIdx.x;

  // stage q[b,:,h,:]
  for (int it = 0; it < 16; ++it) {
    int idx = it * 256 + tid;
    int m = idx >> 6, dk = idx & 63;
    qs[idx] = q[(long)(b * 64 + m) * 512 + h * 64 + dk];
  }
  __syncthreads();

  // scores: so[m][n] = q_m . k_{m,n} / 8 ;  si[n][m] = q_n . k_{m,n} / 8
  for (int it = 0; it < 16; ++it) {
    int idx = it * 256 + tid;
    int m = idx >> 6, n = idx & 63;
    const uint4* kp =
        (const uint4*)(kbf + (long)((b * 64 + m) * 64 + n) * 512 + h * 64);
    float s0 = 0.f, s1 = 0.f;
#pragma unroll
    for (int jj = 0; jj < 8; ++jj) {
      uint4 p = kp[jj];
      unsigned wd[4] = {p.x, p.y, p.z, p.w};
#pragma unroll
      for (int c = 0; c < 4; ++c) {
        int dk = jj * 8 + c * 2;
        float k0 = bf2f((unsigned short)(wd[c] & 0xFFFFu));
        float k1 = bf2f((unsigned short)(wd[c] >> 16));
        s0 += qs[m * 64 + dk] * k0 + qs[m * 64 + dk + 1] * k1;
        s1 += qs[n * 64 + dk] * k0 + qs[n * 64 + dk + 1] * k1;
      }
    }
    so[m * 64 + n] = s0 * 0.125f;
    si[n * 64 + m] = s1 * 0.125f;
  }
  __syncthreads();

  // row softmax: rows 0-63 of so, rows 0-63 of si
  if (tid < 128) {
    float* row = (tid < 64) ? &so[tid * 64] : &si[(tid - 64) * 64];
    float mx = row[0];
    for (int j = 1; j < 64; ++j) mx = fmaxf(mx, row[j]);
    float s = 0.f;
    for (int j = 0; j < 64; ++j) { float e = __expf(row[j] - mx); row[j] = e; s += e; }
    float inv = 1.f / s;
    for (int j = 0; j < 64; ++j) row[j] *= inv;
  }
  __syncthreads();
  if (tid < 64) dg[tid] = so[tid * 64 + tid];
  __syncthreads();

  // message = out_attn + in_attn - diag*eye  (kept in so, written to ws)
  for (int it = 0; it < 16; ++it) {
    int idx = it * 256 + tid;
    int i = idx >> 6, j = idx & 63;
    float mval = so[idx] + si[idx] - ((i == j) ? dg[i] : 0.f);
    so[idx] = mval;
    msg_out[(long)bh * 4096 + idx] = mval;
  }
  __syncthreads();

  // node_hidden[m][dk] = sum_n message[m][n] * v[b,n,h,dk]
  for (int it = 0; it < 16; ++it) {
    int idx = it * 256 + tid;
    int m = idx >> 6, dk = idx & 63;
    float a = 0.f;
    for (int n = 0; n < 64; ++n)
      a += so[m * 64 + n] * v[(long)(b * 64 + n) * 512 + h * 64 + dk];
    nh_out[(long)(b * 64 + m) * 512 + h * 64 + dk] = a;
  }
}

extern "C" void kernel_launch(void* const* d_in, const int* in_sizes, int n_in,
                              void* d_out, int out_size, void* d_ws, size_t ws_size,
                              hipStream_t stream) {
  (void)in_sizes; (void)n_in; (void)out_size; (void)ws_size;

  const float* query_node = (const float*)d_in[0];
  const float* value_node = (const float*)d_in[1];
  const float* key_edge   = (const float*)d_in[2];
  /* d_in[3] = adj_matrix: its softmax is unused by the outputs */
  const float* Wq = (const float*)d_in[4];  const float* bq = (const float*)d_in[5];
  const float* Wk = (const float*)d_in[6];  const float* bk = (const float*)d_in[7];
  const float* Wv = (const float*)d_in[8];  const float* bv = (const float*)d_in[9];
  const float* Wn = (const float*)d_in[10]; const float* bn = (const float*)d_in[11];
  const float* We = (const float*)d_in[12]; const float* be = (const float*)d_in[13];

  // workspace layout
  char* ws = (char*)d_ws;
  float*          qbuf   = (float*)(ws);                              // 2 MB
  float*          vbuf   = (float*)(ws + (size_t)(2 << 20));          // 2 MB
  float*          nhbuf  = (float*)(ws + (size_t)(4 << 20));          // 2 MB
  float*          msgbuf = (float*)(ws + (size_t)(6 << 20));          // 2 MB
  unsigned short* kproj  = (unsigned short*)(ws + (size_t)(8 << 20)); // 64 MB bf16
  unsigned short* wtq    = (unsigned short*)(ws + (size_t)(72 << 20));          // 512 KB each
  unsigned short* wtk    = (unsigned short*)(ws + (size_t)(72 << 20) + 1*(512u << 10));
  unsigned short* wtv    = (unsigned short*)(ws + (size_t)(72 << 20) + 2*(512u << 10));
  unsigned short* wtn    = (unsigned short*)(ws + (size_t)(72 << 20) + 3*(512u << 10));
  unsigned short* wte    = (unsigned short*)(ws + (size_t)(72 << 20) + 4*(512u << 10));

  float* node_out = (float*)d_out;                        // 16*64*512
  float* edge_out = node_out + (size_t)16 * 64 * 512;     // 16*64*64*512

  dim3 blk(256);
  dim3 tgrid(16, 16);
  // pre-convert weights: f32 [k][n] -> bf16 [n][k] (done once, tiny)
  convert_w_kernel<<<tgrid, blk, 0, stream>>>(Wq, wtq);
  convert_w_kernel<<<tgrid, blk, 0, stream>>>(Wk, wtk);
  convert_w_kernel<<<tgrid, blk, 0, stream>>>(Wv, wtv);
  convert_w_kernel<<<tgrid, blk, 0, stream>>>(Wn, wtn);
  convert_w_kernel<<<tgrid, blk, 0, stream>>>(We, wte);

  // q / v projections
  gemm512_wmma<false, false, false><<<16, blk, 0, stream>>>(
      query_node, nullptr, nullptr, wtq, bq, qbuf, nullptr, 16 * 64);
  gemm512_wmma<false, false, false><<<16, blk, 0, stream>>>(
      value_node, nullptr, nullptr, wtv, bv, vbuf, nullptr, 16 * 64);
  // k projection (big GEMM, bf16 output kept in L2)
  gemm512_wmma<false, false, true><<<1024, blk, 0, stream>>>(
      key_edge, nullptr, nullptr, wtk, bk, nullptr, kproj, 16 * 64 * 64);
  // attention: scores -> softmax -> message -> node_hidden
  attn_msg_kernel<<<128, blk, 0, stream>>>(qbuf, vbuf, kproj, msgbuf, nhbuf);
  // node output with mish
  gemm512_wmma<false, true, false><<<16, blk, 0, stream>>>(
      nhbuf, nullptr, nullptr, wtn, bn, node_out, nullptr, 16 * 64);
  // edge output: (message .* k) @ We with mish, edge_hidden never materialized
  gemm512_wmma<true, true, false><<<1024, blk, 0, stream>>>(
      nullptr, kproj, msgbuf, wte, be, edge_out, nullptr, 16 * 64 * 64);
}
